// PrefillWrapperEager_5446018531739
// MI455X (gfx1250) — matrix operand order repaired and checked by reading, not compiled, not run on hardware
//
#include <hip/hip_runtime.h>
#include <hip/hip_bf16.h>

typedef __attribute__((ext_vector_type(16))) _Float16 v16h;
typedef __attribute__((ext_vector_type(8)))  _Float16 v8h;
typedef __attribute__((ext_vector_type(8)))  float    v8f;

#define B_      2
#define S_      2048
#define HQ_     32
#define HKV_    8
#define D_      128
#define W_      512
#define STRIDE_ 32
#define NB_     (S_ / STRIDE_)          // 64
#define ATTN_ELEMS (B_ * S_ * HQ_ * D_) // 16777216
#define SCALE_  0.08838834764831845f    // 1/sqrt(128)
#define MASKED_ (-1.0e30f)

// ---------------------------------------------------------------------------
// Kernel 1: write prefill K/V into the paged cache.
// cache layout: [page][2][HKV][STRIDE][D], page = seq_block_ids[b][ib].
// Pure bandwidth: float4 loads/stores, fully coalesced along D.
// ---------------------------------------------------------------------------
__global__ __launch_bounds__(256)
void cache_write_kernel(const float* __restrict__ k, const float* __restrict__ v,
                        const int* __restrict__ ids, float* __restrict__ cache)
{
    int tid = blockIdx.x * blockDim.x + threadIdx.x;
    int d4 = tid & 31;            // float4 index within D (128/4 = 32)
    int t  = (tid >> 5) & 31;     // token within page
    int hk = (tid >> 10) & 7;     // kv head
    int ib = (tid >> 13) & 63;    // page within sequence
    int bb = tid >> 19;           // batch
    if (bb >= B_) return;

    int page = ids[bb * NB_ + ib];
    size_t src_row = (((size_t)bb * S_ + ib * STRIDE_ + t) * HKV_ + hk) * D_;
    const float4* ks = (const float4*)(k + src_row) + d4;
    const float4* vs = (const float4*)(v + src_row) + d4;
    float4* kd = (float4*)(cache + ((((size_t)page * 2 + 0) * HKV_ + hk) * STRIDE_ + t) * D_) + d4;
    float4* vd = (float4*)(cache + ((((size_t)page * 2 + 1) * HKV_ + hk) * STRIDE_ + t) * D_) + d4;
    *kd = *ks;
    *vd = *vs;
}

// ---------------------------------------------------------------------------
// Kernel 2: sliding-window causal GQA flash attention with sink logit.
// 1 workgroup = 4 waves = 128 threads handles (batch, q-head, 64-query tile).
// Each wave owns 16 queries (one WMMA M-tile). K/V chunks of 32 keys staged
// in LDS as f16; QK^T and PV run on v_wmma_f32_16x16x32_f16 with f32 accum.
// ---------------------------------------------------------------------------
__global__ __launch_bounds__(128)
void attn_kernel(const float* __restrict__ q, const float* __restrict__ k,
                 const float* __restrict__ v, const float* __restrict__ sink,
                 float* __restrict__ out)
{
    __shared__ _Float16 Ksh[STRIDE_][D_];      // 8 KB: K chunk, row-major
    __shared__ _Float16 Vt[D_][STRIDE_];       // 8 KB: V chunk, transposed [d][key]
    __shared__ _Float16 Pw[4][16][32];         // 4 KB: per-wave P tile (C->A relayout)

    const int bid = blockIdx.x;
    const int qt  = bid & 31;          // S/64 = 32 query tiles
    const int h   = (bid >> 5) & 31;   // q head
    const int bb  = bid >> 10;         // batch
    const int hk  = h >> 2;            // GQA: kv head = h / 4

    const int lane = threadIdx.x & 31;
    const int wave = threadIdx.x >> 5;
    const int lo = lane & 15;
    const int hi = lane >> 4;

    const int q0w = qt * 64 + wave * 16;   // first query of this wave's tile

    // ---- load Q fragments once (A-matrix layout: M = lo; K split by half) ----
    const float* qp = q + (((size_t)(bb * S_ + q0w + lo) * HQ_ + h) * D_);
    v16h qa[4];
#pragma unroll
    for (int c = 0; c < 4; ++c) {
        const int d0 = 32 * c;
#pragma unroll
        for (int e = 0; e < 8; ++e) {
            qa[c][e]     = (_Float16)qp[d0 + 16 * hi + e];
            qa[c][8 + e] = (_Float16)qp[d0 + 16 * hi + 8 + e];
        }
    }

    // ---- flash-attention state (C-layout: VGPR r <-> row r + 8*hi) ----
    v8f zero = {};
    v8f o[8];
#pragma unroll
    for (int t = 0; t < 8; ++t) o[t] = zero;
    float m[8], ssum[8];
#pragma unroll
    for (int r = 0; r < 8; ++r) { m[r] = MASKED_; ssum[r] = 0.0f; }

    const int klo  = (qt * 64 - (W_ - 1)) > 0 ? (qt * 64 - (W_ - 1)) : 0;
    const int c0   = klo >> 5;
    const int nch  = ((qt * 64 + 64) >> 5) - c0;

    for (int cc = 0; cc < nch; ++cc) {
        const int ck = (c0 + cc) << 5;    // first key of this 32-key chunk
        __syncthreads();
        // ---- cooperative staging: K row-major, V transposed, f32 -> f16 ----
        {
            const int row = threadIdx.x & 31;
            const int d0  = (threadIdx.x >> 5) * 32;
            const size_t base = (((size_t)bb * S_ + ck + row) * HKV_ + hk) * D_ + d0;
            const float* kp = k + base;
            const float* vp = v + base;
#pragma unroll
            for (int i = 0; i < 32; ++i) {
                Ksh[row][d0 + i] = (_Float16)kp[i];
                Vt[d0 + i][row]  = (_Float16)vp[i];
            }
        }
        __syncthreads();

        // ---- scores: two 16x16 tiles over 32 keys, K-dim = 128 in 4 steps ----
        v8f acc0 = zero, acc1 = zero;
#pragma unroll
        for (int c = 0; c < 4; ++c) {
            const int d0 = 32 * c + 16 * hi;
            v16h b0 = *(const v16h*)&Ksh[lo][d0];        // B-frag: N=key=lo
            v16h b1 = *(const v16h*)&Ksh[16 + lo][d0];   // B-frag: N=key=16+lo
            acc0 = __builtin_amdgcn_wmma_f32_16x16x32_f16(false, qa[c], false, b0,
                                                          (short)0, acc0, false, false);
            acc1 = __builtin_amdgcn_wmma_f32_16x16x32_f16(false, qa[c], false, b1,
                                                          (short)0, acc1, false, false);
        }

        // ---- mask + online softmax (per accumulator row r + 8*hi) ----
        float alpha[8];
#pragma unroll
        for (int r = 0; r < 8; ++r) {
            const int i  = q0w + r + 8 * hi;
            const int j0 = ck + lo;
            const int j1 = ck + 16 + lo;
            float s0 = ((j0 <= i) && (i - j0 < W_)) ? acc0[r] * SCALE_ : MASKED_;
            float s1 = ((j1 <= i) && (i - j1 < W_)) ? acc1[r] * SCALE_ : MASKED_;
            float mt = fmaxf(s0, s1);
            mt = fmaxf(mt, __shfl_xor(mt, 1, 32));
            mt = fmaxf(mt, __shfl_xor(mt, 2, 32));
            mt = fmaxf(mt, __shfl_xor(mt, 4, 32));
            mt = fmaxf(mt, __shfl_xor(mt, 8, 32));
            const float mn = fmaxf(m[r], mt);
            const float al = __expf(m[r] - mn);
            const float p0 = __expf(s0 - mn);
            const float p1 = __expf(s1 - mn);
            float rs = p0 + p1;
            rs += __shfl_xor(rs, 1, 32);
            rs += __shfl_xor(rs, 2, 32);
            rs += __shfl_xor(rs, 4, 32);
            rs += __shfl_xor(rs, 8, 32);
            ssum[r] = ssum[r] * al + rs;
            m[r] = mn;
            alpha[r] = al;
            // stash P in natural [M][N] order for the A-layout reload
            Pw[wave][r + 8 * hi][lo]      = (_Float16)p0;
            Pw[wave][r + 8 * hi][16 + lo] = (_Float16)p1;
        }

        // ---- reload P as A-fragment (M = lo, K = key-in-chunk) ----
        v16h pa;
        {
            v8h a = *(const v8h*)&Pw[wave][lo][16 * hi];
            v8h b = *(const v8h*)&Pw[wave][lo][16 * hi + 8];
#pragma unroll
            for (int e = 0; e < 8; ++e) { pa[e] = a[e]; pa[8 + e] = b[e]; }
        }

        // ---- PV: 8 output d-tiles, rescale accumulator rows then wmma ----
#pragma unroll
        for (int t = 0; t < 8; ++t) {
            v8f ot = o[t];
#pragma unroll
            for (int r = 0; r < 8; ++r) ot[r] *= alpha[r];
            v16h bv = *(const v16h*)&Vt[16 * t + lo][16 * hi];  // B-frag: N=d, K=key
            o[t] = __builtin_amdgcn_wmma_f32_16x16x32_f16(false, pa, false, bv,
                                                          (short)0, ot, false, false);
        }
    }

    // ---- fold in sink logit (extra softmax column, dropped from output) ----
    const float sk = sink[h];
    float den[8], alf[8];
#pragma unroll
    for (int r = 0; r < 8; ++r) {
        const float mn = fmaxf(m[r], sk);
        const float al = __expf(m[r] - mn);
        den[r] = ssum[r] * al + __expf(sk - mn);
        alf[r] = al;
    }
#pragma unroll
    for (int t = 0; t < 8; ++t) {
#pragma unroll
        for (int r = 0; r < 8; ++r) {
            size_t idx = (((size_t)bb * S_ + q0w + r + 8 * hi) * HQ_ + h) * D_ + 16 * t + lo;
            out[idx] = o[t][r] * alf[r] / den[r];
        }
    }
}

// ---------------------------------------------------------------------------
extern "C" void kernel_launch(void* const* d_in, const int* in_sizes, int n_in,
                              void* d_out, int out_size, void* d_ws, size_t ws_size,
                              hipStream_t stream)
{
    (void)in_sizes; (void)n_in; (void)out_size; (void)d_ws; (void)ws_size;
    const float* q    = (const float*)d_in[0];
    const float* k    = (const float*)d_in[1];
    const float* v    = (const float*)d_in[2];
    // d_in[3] = cache_state (zeros; every page slot is overwritten)
    const int*   ids  = (const int*)d_in[4];
    const float* sink = (const float*)d_in[5];

    float* out   = (float*)d_out;
    float* cache = out + ATTN_ELEMS;

    // B*NB*HKV*32 rows * 32 float4 = 1,048,576 threads
    cache_write_kernel<<<4096, 256, 0, stream>>>(k, v, ids, cache);
    // B * HQ * (S/64) = 2048 workgroups of 4 waves
    attn_kernel<<<B_ * HQ_ * (S_ / 64), 128, 0, stream>>>(q, k, v, sink, out);
}
